// SB_Layer_21895743275225
// MI455X (gfx1250) — matrix operand order repaired and checked by reading, not compiled, not run on hardware
//
#include <hip/hip_runtime.h>
#include <hip/hip_bf16.h>
#include <stdint.h>

// ---------------- types ----------------
typedef __attribute__((ext_vector_type(8)))  float          v8f;
typedef __attribute__((ext_vector_type(16))) __bf16         v16bf;
typedef __attribute__((ext_vector_type(16))) unsigned short v16u;
typedef __attribute__((ext_vector_type(8)))  unsigned short v8u;

#define B_DIM 16384
#define D_DIM 1024
#define K_DIM 512
#define N_DIM 1024   // K*U
#define INV_TEMP (1.0f/0.67f)

// ---------------- helpers ----------------
__device__ __forceinline__ unsigned hash32(unsigned x) {
  x ^= x >> 16; x *= 0x7feb352dU;
  x ^= x >> 15; x *= 0x846ca68bU;
  x ^= x >> 16; return x;
}
__device__ __forceinline__ float u01(unsigned h) {
  float u = (float)(h >> 8) * (1.0f / 16777216.0f);
  return fminf(fmaxf(u, 1e-6f), 1.0f - 1e-6f);
}
__device__ __forceinline__ float softplusf(float x) {
  return (x > 20.f) ? x : log1pf(expf(x));
}
__device__ __forceinline__ unsigned short bf16_rne(float f) {
  unsigned x = __float_as_uint(f);
  unsigned r = x + 0x7fffU + ((x >> 16) & 1U);
  return (unsigned short)(r >> 16);
}
__device__ __forceinline__ float digammaf_(float x) {
  float r = 0.f;
  while (x < 6.f) { r -= 1.f / x; x += 1.f; }
  float f = 1.f / (x * x);
  r += logf(x) - 0.5f / x
     - f * (1.f/12.f - f * (1.f/120.f - f * (1.f/252.f)));
  return r;
}

// A fragment (bf16 row-major -> WMMA 16x32 A layout):
// lane<16: K = {kk..kk+7, kk+16..kk+23}; lane>=16: shifted by 8.
// p = Ab + row*D + (lane>=16 ? 8 : 0) + kk  -> two 16B loads, gap 16 elems
__device__ __forceinline__ v16bf load_afrag(const unsigned short* __restrict__ p) {
  v8u lo = *(const v8u*)(p + 0);
  v8u hi = *(const v8u*)(p + 16);
  v16u t;
#pragma unroll
  for (int i = 0; i < 8; ++i) { t[i] = lo[i]; t[8 + i] = hi[i]; }
  return __builtin_bit_cast(v16bf, t);
}

// B fragment (WeffT[n][d] bf16 row-major -> WMMA 32x16 B layout):
// lane<16: K = {kk..kk+7, kk+16..kk+23}; lane>=16: shifted by 8
__device__ __forceinline__ v16bf load_bfrag(const unsigned short* __restrict__ p) {
  v8u lo = *(const v8u*)(p + 0);
  v8u hi = *(const v8u*)(p + 16);
  v16u t;
#pragma unroll
  for (int i = 0; i < 8; ++i) { t[i] = lo[i]; t[8 + i] = hi[i]; }
  return __builtin_bit_cast(v16bf, t);
}

// ---------------- kernel 1: kl_sticks (initializes loss) ----------------
__global__ __launch_bounds__(512) void k_sticks(const float* __restrict__ conc1,
                                                const float* __restrict__ conc0,
                                                float* __restrict__ loss) {
  int k = threadIdx.x;
  float a = softplusf(conc1[k]);
  float b = softplusf(conc0[k]);
  float term = ((a - 1.f) / a) * (-0.57721566f - digammaf_(b) - 1.f / b)
             + logf(a * b + 1e-8f) - (b - 1.f) / b;
  __shared__ float sm[512];
  sm[k] = term; __syncthreads();
  for (int s = 256; s > 0; s >>= 1) {
    if (k < s) sm[k] += sm[k + s];
    __syncthreads();
  }
  if (k == 0) loss[0] = sm[0];   // first writer: init, not atomic
}

// ---------------- kernel 2: z_sample + cumprod prior + kl_z ----------------
__global__ __launch_bounds__(256) void k_zcol(const float* __restrict__ t_pi,
                                              const float* __restrict__ conc1,
                                              const float* __restrict__ conc0,
                                              float* __restrict__ zs,
                                              float* __restrict__ loss) {
  int k = blockIdx.x * 256 + threadIdx.x;   // 0..511, one column each
  float a = softplusf(conc1[k]);
  float b = softplusf(conc0[k]);
  float inva = 1.f / a, invb = 1.f / b;
  float cum = 1.f, kl = 0.f;
  for (int d = 0; d < D_DIM; ++d) {
    unsigned i = (unsigned)(d * K_DIM + k);
    // relaxed-Bernoulli sample of t_pi
    float u1 = u01(hash32(i ^ 0x27d4eb2fU));
    float l  = logf(u1) - log1pf(-u1);
    float tp = t_pi[(size_t)d * K_DIM + k];
    float z  = 1.f / (1.f + expf(-(tp + l) * INV_TEMP));
    zs[(size_t)d * K_DIM + k] = z;
    // Kumaraswamy stick-breaking prior
    float uk = u01(hash32(i ^ 0x165667b1U));
    float qu = powf(1.f - powf(uk, invb), inva);
    cum *= qu;
    float q = 1.f / (1.f + expf(-tp));
    kl += q * (logf(q + 1e-8f) - logf(cum + 1e-8f));
  }
  atomicAdd(loss, kl);
}

// ---------------- kernel 3: Weff^T (bf16) + kl_weights ----------------
__global__ __launch_bounds__(256) void k_weights(const float* __restrict__ mW,
                                                 const float* __restrict__ sW,
                                                 const float* __restrict__ zs,
                                                 unsigned short* __restrict__ WT,
                                                 float* __restrict__ loss) {
  int gid = blockIdx.x * 256 + threadIdx.x;   // 0 .. 1048575
  int n = gid >> 10;
  int d = gid & (D_DIM - 1);
  size_t src = (size_t)d * N_DIM + n;
  float m = mW[src], sv = sW[src];
  float sp = softplusf(sv);
  unsigned i = (unsigned)src;
  float u1 = u01(hash32(i ^ 0x85ebca6bU));
  float u2 = u01(hash32(i ^ 0xc2b2ae35U));
  float eps = sqrtf(-2.f * logf(u1)) * cosf(6.28318530718f * u2);
  float W = m + eps * sp;
  float z = zs[(size_t)d * K_DIM + (n & (K_DIM - 1))];   // tile(z,(1,U))
  WT[(size_t)n * D_DIM + d] = bf16_rne(z * W);           // transposed store
  float t = 2.f * sp - m * m - sp * sp + 1.f;
  __shared__ float sm[256];
  sm[threadIdx.x] = t; __syncthreads();
  for (int s = 128; s > 0; s >>= 1) {
    if (threadIdx.x < s) sm[threadIdx.x] += sm[threadIdx.x + s];
    __syncthreads();
  }
  if (threadIdx.x == 0)
    atomicAdd(loss, sm[0] * (-0.5f / (1048576.0f * 60000.0f)));
}

// ---------------- kernel 3b: one-time f32 -> bf16 conversion of inputs ----
// 16M elements, 8 per thread; removes all conversion VALU from the GEMM loop
__global__ __launch_bounds__(256) void k_convA(const float* __restrict__ A,
                                               unsigned short* __restrict__ Ab) {
  size_t base = ((size_t)blockIdx.x * 256 + threadIdx.x) * 8;
  float4 f0 = *(const float4*)(A + base);
  float4 f1 = *(const float4*)(A + base + 4);
  v8u o;
  o[0] = bf16_rne(f0.x); o[1] = bf16_rne(f0.y);
  o[2] = bf16_rne(f0.z); o[3] = bf16_rne(f0.w);
  o[4] = bf16_rne(f1.x); o[5] = bf16_rne(f1.y);
  o[6] = bf16_rne(f1.z); o[7] = bf16_rne(f1.w);
  *(v8u*)(Ab + base) = o;
}

// ---------------- kernel 4: WMMA GEMM + fused LWTA + kl_xi ----------------
// block = 256 threads = 8 waves arranged 2(M) x 4(N) -> 128x128 block tile
// wave = 64x32 outputs = 4x2 WMMA f32 16x16 tiles, K stepped by 32 (bf16)
// inner loop: 12x 16B loads : 8 wmma per K-step, zero conversion VALU
__global__ __launch_bounds__(256) void k_gemm_lwta(const unsigned short* __restrict__ Ab,
                                                   const unsigned short* __restrict__ WT,
                                                   const float* __restrict__ bias,
                                                   float* __restrict__ out,
                                                   float* __restrict__ loss) {
  const int tid  = threadIdx.x;
  const int lane = tid & 31;
  const int w    = tid >> 5;
  const int wm   = w & 1;        // 2 waves along M
  const int wn   = w >> 1;       // 4 waves along N
  const int Mb   = blockIdx.y * 128 + wm * 64;
  const int Nb   = blockIdx.x * 128 + wn * 32;
  const int hi   = (lane >> 4) & 1;
  const int r15  = lane & 15;

  v8f acc[4][2] = {};

  const unsigned short* ap[4];
  const unsigned short* bp[2];
#pragma unroll
  for (int ti = 0; ti < 4; ++ti)
    ap[ti] = Ab + (size_t)(Mb + ti * 16 + r15) * D_DIM + hi * 8;
#pragma unroll
  for (int tj = 0; tj < 2; ++tj)
    bp[tj] = WT + (size_t)(Nb + tj * 16 + r15) * D_DIM + hi * 8;

#pragma unroll 2
  for (int kk = 0; kk < D_DIM; kk += 32) {
    v16bf af[4], bfr[2];
#pragma unroll
    for (int ti = 0; ti < 4; ++ti) {
      __builtin_prefetch(ap[ti] + kk + 64, 0, 0);   // global_prefetch_b8
      af[ti] = load_afrag(ap[ti] + kk);
    }
#pragma unroll
    for (int tj = 0; tj < 2; ++tj) bfr[tj] = load_bfrag(bp[tj] + kk);
#pragma unroll
    for (int ti = 0; ti < 4; ++ti)
#pragma unroll
      for (int tj = 0; tj < 2; ++tj)
        acc[ti][tj] = __builtin_amdgcn_wmma_f32_16x16x32_bf16(
            false, af[ti], false, bfr[tj], (short)0, acc[ti][tj], false, false);
  }

  // fused LWTA epilogue: partner column N^1 lives in lane^1
  float kl = 0.f;
  float bv[2];
#pragma unroll
  for (int tj = 0; tj < 2; ++tj) bv[tj] = bias[Nb + tj * 16 + r15];

#pragma unroll
  for (int ti = 0; ti < 4; ++ti) {
#pragma unroll
    for (int tj = 0; tj < 2; ++tj) {
      v8f c = acc[ti][tj];
      int N = Nb + tj * 16 + r15;
#pragma unroll
      for (int r = 0; r < 8; ++r) {
        int M = Mb + ti * 16 + hi * 8 + r;
        float a  = c[r] + bv[tj];
        float ap2 = __shfl_xor(a, 1, 32);
        unsigned h = hash32((unsigned)(M * N_DIM + N) ^ 0x3c6ef372U);
        float g  = -logf(-logf(u01(h)));
        float gp = __shfl_xor(g, 1, 32);
        float xi = 1.f / (1.f + expf((ap2 + gp - a - g) * INV_TEMP));
        out[(size_t)M * N_DIM + N] = a * xi;
        float q = 1.f / (1.f + expf(ap2 - a));      // 2-way softmax (no gumbel)
        kl += q * (logf(q + 1e-8f) + 0.69314718f);  // - log(1/U), U=2
      }
    }
  }
  for (int o = 16; o > 0; o >>= 1) kl += __shfl_down(kl, o, 32);
  __shared__ float sm[8];
  if (lane == 0) sm[w] = kl;
  __syncthreads();
  if (tid == 0) {
    float s = 0.f;
#pragma unroll
    for (int i = 0; i < 8; ++i) s += sm[i];
    atomicAdd(loss, s * (1.0f / (16384.0f * 2.0f)));  // mean over B*U
  }
}

// ---------------- host launcher ----------------
extern "C" void kernel_launch(void* const* d_in, const int* in_sizes, int n_in,
                              void* d_out, int out_size, void* d_ws, size_t ws_size,
                              hipStream_t stream) {
  (void)in_sizes; (void)n_in; (void)out_size; (void)ws_size;
  const float* inputs = (const float*)d_in[0];
  const float* mW     = (const float*)d_in[1];
  const float* sW     = (const float*)d_in[2];
  const float* conc1  = (const float*)d_in[3];
  const float* conc0  = (const float*)d_in[4];
  const float* t_pi   = (const float*)d_in[5];
  const float* biases = (const float*)d_in[6];

  float* out  = (float*)d_out;
  float* loss = out + (size_t)B_DIM * N_DIM;     // scalar appended after out_act

  // workspace: [0,32MB) bf16 A, [32MB,34MB) bf16 Weff^T, [34MB,36MB) z_sample
  unsigned short* Ab = (unsigned short*)d_ws;
  unsigned short* WT = (unsigned short*)((char*)d_ws + (size_t)32 * 1024 * 1024);
  float*          zs = (float*)((char*)d_ws + (size_t)34 * 1024 * 1024);

  k_sticks <<<1,                    512, 0, stream>>>(conc1, conc0, loss);
  k_zcol   <<<2,                    256, 0, stream>>>(t_pi, conc1, conc0, zs, loss);
  k_weights<<<(D_DIM*N_DIM)/256,    256, 0, stream>>>(mW, sW, zs, WT, loss);
  k_convA  <<<(B_DIM*D_DIM)/(256*8),256, 0, stream>>>(inputs, Ab);
  k_gemm_lwta<<<dim3(N_DIM/128, B_DIM/128), 256, 0, stream>>>(Ab, WT, biases, out, loss);
}
